// ConnectNet_19370302505104
// MI455X (gfx1250) — matrix operand order repaired and checked by reading, not compiled, not run on hardware
//
#include <hip/hip_runtime.h>

typedef __attribute__((ext_vector_type(16))) __bf16 v16bf;
typedef __attribute__((ext_vector_type(8)))  float  v8f;

#define C_IN   64
#define H_IN   224
#define W_IN   224
#define HW_IN  (H_IN * W_IN)
#define C_OUT  128
#define H_OUT  222
#define W_OUT  222
#define NKB    18                         // K blocks of 32 (K total = 576)
#define NX     (C_IN * H_IN * W_IN)       // 3211264 (divisible by 4)
#define AW_ELEMS (NKB * 8 * 32 * 16)      // 73728 bf16 = 147456 B
#define AW_BYTES (AW_ELEMS * 2)
#define BROW   40                         // LDS B row: 32 bf16 + 8 pad (80 B)

#if defined(__gfx1250__) && __has_builtin(__builtin_amdgcn_global_load_async_to_lds_b128)
#define HAVE_ASYNC_LDS 1
typedef int v4i_vs __attribute__((vector_size(16)));
typedef __attribute__((address_space(1))) v4i_vs* gv4i_p;
typedef __attribute__((address_space(3))) v4i_vs* lv4i_p;
#else
#define HAVE_ASYNC_LDS 0
#endif

__device__ __forceinline__ unsigned short f2bf(float f) {
  union { float f; unsigned u; } v; v.f = f;
  unsigned u = v.u;
  return (unsigned short)((u + 0x7FFFu + ((u >> 16) & 1u)) >> 16);  // RNE
}

// Pack OIHW f32 weights into kb-major per-lane WMMA A-fragment bf16 layout:
// flat = ((kb*8 + mt)*32 + lane)*16 + h
// lane: M = mt*16 + (lane&15); K = kb*32 + (lane>>4)*16 + h
// K -> (tap r = K>>6, channel c = K&63); weight index = m*576 + c*9 + r
__global__ void pack_weights_bf16(const float* __restrict__ w,
                                  unsigned short* __restrict__ aw) {
  int idx = blockIdx.x * 256 + threadIdx.x;
  if (idx >= AW_ELEMS) return;
  int h  = idx & 15;
  int l  = (idx >> 4) & 31;
  int t  = idx >> 9;
  int mt = t & 7;
  int kb = t >> 3;
  int m  = mt * 16 + (l & 15);
  int K  = kb * 32 + ((l >> 4) * 16) + h;
  int r  = K >> 6;
  int c  = K & 63;
  aw[idx] = f2bf(w[m * (C_IN * 9) + c * 9 + r]);
}

// One-pass f32 -> bf16 conversion of x (vectorized: 4 elems/thread, exact fit)
__global__ void convert_x_bf16(const float* __restrict__ x,
                               unsigned short* __restrict__ xb) {
  int i = (blockIdx.x * 256 + threadIdx.x) * 4;
  float4 v = *(const float4*)(x + i);
  uint2 p;
  p.x = (unsigned)f2bf(v.x) | ((unsigned)f2bf(v.y) << 16);
  p.y = (unsigned)f2bf(v.z) | ((unsigned)f2bf(v.w) << 16);
  *(uint2*)(xb + i) = p;
}

__global__ __launch_bounds__(256) void conv3x3_wmma(
    const unsigned short* __restrict__ xb,
    const unsigned short* __restrict__ aw,
    float* __restrict__ out) {
  __shared__ unsigned short Ab[2][8 * 32 * 16];   // 2 x 8KB: per-kb A fragments
  __shared__ unsigned short Bb[2][128 * BROW];    // 2 x 10KB: 32K x 128N bf16

  const int tid = threadIdx.x;
  const int oh  = blockIdx.y;
  const int ow0 = blockIdx.x * 128;

  const int l  = tid & 31;
  const int wv = tid >> 5;
  const int mq = wv >> 1;     // m quarter: 2 m-tiles (32 rows)
  const int ng = wv & 1;      // n half: 4 n-tiles (64 cols)

  // B staging role: 16 consecutive channels (K) for one pixel (n)
  const int bn  = tid >> 1;          // 0..127
  const int bk0 = (tid & 1) * 16;    // 0 or 16
  int own = ow0 + bn;
  if (own > W_OUT - 1) own = W_OUT - 1;   // clamp (computed, never stored)

  v8f acc[2][4] = {};

  auto stage = [&](int kb, int buf) {
    // ---- A tile: contiguous 8KB chunk, 32B per thread, async global->LDS ----
    const unsigned short* ag = aw + (size_t)kb * 4096 + tid * 16;
    unsigned short* al = &Ab[buf][tid * 16];
#if HAVE_ASYNC_LDS
    __builtin_amdgcn_global_load_async_to_lds_b128((gv4i_p)ag, (lv4i_p)al, 0, 0);
    __builtin_amdgcn_global_load_async_to_lds_b128((gv4i_p)ag, (lv4i_p)al, 16, 0);
#else
    *(uint4*)al       = *(const uint4*)ag;
    *(uint4*)(al + 8) = *(const uint4*)(ag + 8);
#endif
    // ---- B tile: im2col gather of 16 channels (bf16) for one pixel ----
    const int r  = kb >> 1;           // tap 0..8 (K = r*64 + c)
    const int c0 = (kb & 1) * 32;
    const int kh = r / 3, kw = r - 3 * kh;
    const unsigned short* src = xb + (size_t)(c0 + bk0) * HW_IN
                                   + (size_t)(oh + kh) * W_IN + (own + kw);
    unsigned pk[8];
#pragma unroll
    for (int i = 0; i < 8; ++i) {
      unsigned lo = src[(size_t)(2 * i)     * HW_IN];
      unsigned hi = src[(size_t)(2 * i + 1) * HW_IN];
      pk[i] = lo | (hi << 16);
    }
    uint4* d = (uint4*)&Bb[buf][bn * BROW + bk0];
    d[0] = make_uint4(pk[0], pk[1], pk[2], pk[3]);
    d[1] = make_uint4(pk[4], pk[5], pk[6], pk[7]);
  };

  stage(0, 0);
  asm volatile("s_wait_asynccnt 0" ::: "memory");
  __syncthreads();

#pragma unroll 1
  for (int kb = 0; kb < NKB; ++kb) {
    if (kb + 1 < NKB) stage(kb + 1, (kb + 1) & 1);

    // B fragments: lane l -> column n = ng*64 + j*16 + (l&15), K = (l>>4)*16 + h
    v16bf bfr[4];
#pragma unroll
    for (int j = 0; j < 4; ++j) {
      const unsigned short* bp =
          &Bb[kb & 1][(ng * 64 + j * 16 + (l & 15)) * BROW + (l >> 4) * 16];
      union { uint4 u[2]; v16bf v; } t;
      t.u[0] = *(const uint4*)bp;
      t.u[1] = *(const uint4*)(bp + 8);
      bfr[j] = t.v;
    }
    // A fragments from LDS: lane-contiguous 32B
    v16bf afr[2];
#pragma unroll
    for (int i = 0; i < 2; ++i) {
      const unsigned short* ap = &Ab[kb & 1][((mq * 2 + i) * 32 + l) * 16];
      union { uint4 u[2]; v16bf v; } t;
      t.u[0] = *(const uint4*)ap;
      t.u[1] = *(const uint4*)(ap + 8);
      afr[i] = t.v;
    }
#pragma unroll
    for (int i = 0; i < 2; ++i)
#pragma unroll
      for (int j = 0; j < 4; ++j)
        acc[i][j] = __builtin_amdgcn_wmma_f32_16x16x32_bf16(
            false, afr[i], false, bfr[j], (short)0, acc[i][j], false, false);

    asm volatile("s_wait_asynccnt 0" ::: "memory");
    __syncthreads();
  }

  // Epilogue: C/D layout — lane l holds N=l&15; rows M = hi*8 + vgpr
  const int hi = l >> 4;
#pragma unroll
  for (int j = 0; j < 4; ++j) {
    const int ow = ow0 + ng * 64 + j * 16 + (l & 15);
    if (ow < W_OUT) {
#pragma unroll
      for (int i = 0; i < 2; ++i)
#pragma unroll
        for (int r2 = 0; r2 < 8; ++r2) {
          int m = mq * 32 + i * 16 + hi * 8 + r2;
          out[(size_t)m * (H_OUT * W_OUT) + (size_t)oh * W_OUT + ow] =
              acc[i][j][r2];
        }
    }
  }
}

extern "C" void kernel_launch(void* const* d_in, const int* in_sizes, int n_in,
                              void* d_out, int out_size, void* d_ws, size_t ws_size,
                              hipStream_t stream) {
  const float* x = (const float*)d_in[0];       // (64,224,224) f32
  const float* w = (const float*)d_in[1];       // (128,64,3,3) f32
  float* out = (float*)d_out;                   // (128,222,222) f32

  unsigned short* aw = (unsigned short*)d_ws;                     // 147456 B
  unsigned short* xbf = (unsigned short*)((char*)d_ws + AW_BYTES); // 6422528 B

  pack_weights_bf16<<<(AW_ELEMS + 255) / 256, 256, 0, stream>>>(w, aw);
  convert_x_bf16<<<NX / 4 / 256, 256, 0, stream>>>(x, xbf);

  dim3 grid((W_OUT + 127) / 128, H_OUT, 1);     // 2 x 222 blocks
  conv3x3_wmma<<<grid, 256, 0, stream>>>(xbf, aw, out);
}
